// MultiheadLatentAttention_79319456022756
// MI455X (gfx1250) — compile-verified, matrix operand order checked by reading, not run on hardware
//
#include <hip/hip_runtime.h>
#include <hip/hip_bf16.h>
#include <stdint.h>

// ---------------------------------------------------------------------------
// MLA forward for MI455X (gfx1250, wave32, WMMA bf16 16x16x32)
// B=2, L=2048, D=1024, H=16, DH=64, DC=512
// ---------------------------------------------------------------------------

#define MLA_B  2
#define MLA_L  2048
#define MLA_D  1024
#define MLA_H  16
#define MLA_DH 64
#define MLA_DC 512

typedef __attribute__((ext_vector_type(16))) __bf16 v16bf;
typedef __attribute__((ext_vector_type(8)))  float  v8f;

struct B32x8 { uint4 lo, hi; };   // 32 bytes = 16 bf16, matches v16bf

static __device__ inline v8f wmma_bf16(const B32x8& a, const B32x8& b, v8f c) {
    v16bf av = __builtin_bit_cast(v16bf, a);
    v16bf bv = __builtin_bit_cast(v16bf, b);
    return __builtin_amdgcn_wmma_f32_16x16x32_bf16(
        /*neg_a=*/false, av, /*neg_b=*/false, bv,
        /*c_mod=*/(short)0, c, /*reuse_a=*/false, /*reuse_b=*/false);
}

// round-to-nearest-even f32 -> bf16 (as raw u16)
static __device__ inline uint16_t f2bf(float f) {
    uint32_t u = __float_as_uint(f);
    u += 0x7FFFu + ((u >> 16) & 1u);
    return (uint16_t)(u >> 16);
}

// reductions within a 16-lane half of the wave (xor masks < 16 stay in half)
static __device__ inline float half16_max(float v) {
    #pragma unroll
    for (int m = 1; m < 16; m <<= 1) v = fmaxf(v, __shfl_xor(v, m, 32));
    return v;
}
static __device__ inline float half16_sum(float v) {
    #pragma unroll
    for (int m = 1; m < 16; m <<= 1) v += __shfl_xor(v, m, 32);
    return v;
}

// ---------------------------------------------------------------------------
// elementwise f32 -> bf16
__global__ void k_cast_bf16(const float* __restrict__ in, uint16_t* __restrict__ out, int n) {
    int i = blockIdx.x * blockDim.x + threadIdx.x;
    if (i < n) out[i] = f2bf(in[i]);
}

// W [K,N] f32 -> Wt [N,K] bf16 (so GEMM B-operand loads are contiguous in K)
__global__ void k_cast_transpose_w(const float* __restrict__ W, uint16_t* __restrict__ Wt,
                                   int K, int N) {
    int i = blockIdx.x * blockDim.x + threadIdx.x;
    if (i >= K * N) return;
    int k = i / N, n = i % N;
    Wt[(size_t)n * K + k] = f2bf(W[i]);
}

// v [B*L, D] bf16 -> vt [(B*H), DH, L] bf16
__global__ void k_transpose_v(const uint16_t* __restrict__ v, uint16_t* __restrict__ vt) {
    int i = blockIdx.x * blockDim.x + threadIdx.x;
    const int total = MLA_B * MLA_L * MLA_D;
    if (i >= total) return;
    int bl = i / MLA_D, c = i % MLA_D;
    int b = bl / MLA_L, l = bl % MLA_L;
    int h = c / MLA_DH, d = c % MLA_DH;
    vt[((size_t)(b * MLA_H + h) * MLA_DH + d) * MLA_L + l] = v[i];
}

// ---------------------------------------------------------------------------
// Register-blocked GEMM: C[M,N] = A[M,K](bf16) x Bt[N,K](bf16)^T + bias[N]
// Each wave computes a 32x64 block = 2x4 WMMA tiles; per 32-wide k-step it
// loads 2 A tiles + 4 B tiles (12 x b128) and issues 8 WMMAs (1.5 loads/WMMA).
// A/B operand loads follow the CDNA5 16-bit striping:
//   lanes 0-15: K = kt+0..7 (v0..3) and kt+16..23 (v4..7)
//   lanes 16-31: K = kt+8..15 and kt+24..31
#define GEMM_WAVES 4
#define GEMM_MT 2      // 16-row tiles per wave
#define GEMM_NT 4      // 16-col tiles per wave
__global__ void k_gemm_bf16(const uint16_t* __restrict__ A,
                            const uint16_t* __restrict__ Bt,
                            const float*    __restrict__ bias,
                            void* __restrict__ Cv,
                            int M, int N, int K, int c_is_f32) {
    const int lane = threadIdx.x & 31;
    const int wave = threadIdx.x >> 5;
    const int lm   = lane & 15;
    const int half = lane >> 4;

    const int tilesN = N / (16 * GEMM_NT);
    const int tilesM = M / (16 * GEMM_MT);
    int tile = blockIdx.x * GEMM_WAVES + wave;
    if (tile >= tilesM * tilesN) return;
    const int m0 = (tile / tilesN) * 16 * GEMM_MT;
    const int n0 = (tile % tilesN) * 16 * GEMM_NT;

    const uint16_t* arow[GEMM_MT];
    const uint16_t* bcol[GEMM_NT];
    #pragma unroll
    for (int mi = 0; mi < GEMM_MT; ++mi) arow[mi] = A  + (size_t)(m0 + mi * 16 + lm) * K;
    #pragma unroll
    for (int ni = 0; ni < GEMM_NT; ++ni) bcol[ni] = Bt + (size_t)(n0 + ni * 16 + lm) * K;

    v8f acc[GEMM_MT][GEMM_NT];
    #pragma unroll
    for (int mi = 0; mi < GEMM_MT; ++mi)
        #pragma unroll
        for (int ni = 0; ni < GEMM_NT; ++ni) acc[mi][ni] = (v8f){};

    for (int kt = 0; kt < K; kt += 32) {
        B32x8 a[GEMM_MT], b[GEMM_NT];
        #pragma unroll
        for (int mi = 0; mi < GEMM_MT; ++mi) {
            a[mi].lo = *(const uint4*)(arow[mi] + kt + half * 8);
            a[mi].hi = *(const uint4*)(arow[mi] + kt + 16 + half * 8);
        }
        #pragma unroll
        for (int ni = 0; ni < GEMM_NT; ++ni) {
            b[ni].lo = *(const uint4*)(bcol[ni] + kt + half * 8);
            b[ni].hi = *(const uint4*)(bcol[ni] + kt + 16 + half * 8);
        }
        #pragma unroll
        for (int mi = 0; mi < GEMM_MT; ++mi)
            #pragma unroll
            for (int ni = 0; ni < GEMM_NT; ++ni)
                acc[mi][ni] = wmma_bf16(a[mi], b[ni], acc[mi][ni]);
    }

    #pragma unroll
    for (int ni = 0; ni < GEMM_NT; ++ni) {
        const float bv = bias ? bias[n0 + ni * 16 + lm] : 0.0f;
        #pragma unroll
        for (int mi = 0; mi < GEMM_MT; ++mi) {
            if (c_is_f32) {
                float* C = (float*)Cv;
                #pragma unroll
                for (int r = 0; r < 8; ++r) {
                    int m = m0 + mi * 16 + r + 8 * half;   // C layout: vgpr r -> row r (+8 hi half)
                    C[(size_t)m * N + n0 + ni * 16 + lm] = acc[mi][ni][r] + bv;
                }
            } else {
                uint16_t* C = (uint16_t*)Cv;
                #pragma unroll
                for (int r = 0; r < 8; ++r) {
                    int m = m0 + mi * 16 + r + 8 * half;
                    C[(size_t)m * N + n0 + ni * 16 + lm] = f2bf(acc[mi][ni][r] + bv);
                }
            }
        }
    }
}

// ---------------------------------------------------------------------------
// Flash attention: one wave handles (b, h, 32-query tile) as 2 query sub-tiles
// sharing every loaded K-chunk and V-chunk operand. Loops over keys in
// 32-wide chunks with online softmax. All matmuls via WMMA bf16.
#define ATT_WAVES 4
#define ATT_QS    2        // 16-query sub-tiles per wave
__global__ void k_mla_attention(const uint16_t* __restrict__ qb,   // [B*L, D]
                                const uint16_t* __restrict__ kb,   // [B*L, D]
                                const uint16_t* __restrict__ vt,   // [(B*H), DH, L]
                                const unsigned char* __restrict__ mask, // [B*L]
                                uint16_t* __restrict__ attn)       // [B*L, D]
{
    __shared__ uint16_t Pt[ATT_WAVES][ATT_QS][16 * 32];  // per-wave exp(S) tiles

    const int lane = threadIdx.x & 31;
    const int wave = threadIdx.x >> 5;
    const int lm   = lane & 15;
    const int half = lane >> 4;

    const int qtiles = MLA_L / (16 * ATT_QS);            // 64
    int flat = blockIdx.x * ATT_WAVES + wave;            // 0 .. B*H*qtiles-1
    const int bh = flat / qtiles;
    const int qt = flat % qtiles;
    const int b  = bh / MLA_H;
    const int h  = bh % MLA_H;
    const int q0 = qt * 16 * ATT_QS;

    // Persistent Q operands: A-layout, per sub-tile two 16x32 tiles over DH=64
    B32x8 qa[ATT_QS][2];
    #pragma unroll
    for (int qs = 0; qs < ATT_QS; ++qs) {
        const uint16_t* qrow =
            qb + ((size_t)(b * MLA_L + q0 + qs * 16 + lm) * MLA_D + h * MLA_DH);
        #pragma unroll
        for (int t = 0; t < 2; ++t) {
            qa[qs][t].lo = *(const uint4*)(qrow + t * 32 + half * 8);
            qa[qs][t].hi = *(const uint4*)(qrow + t * 32 + 16 + half * 8);
        }
    }

    const float sc  = 0.125f;                            // 1/sqrt(64)
    const float NEG = -1e30f;

    v8f O[ATT_QS][4];                                    // output accum over DH
    float mrow[ATT_QS][8], lsum[ATT_QS][8];
    #pragma unroll
    for (int qs = 0; qs < ATT_QS; ++qs) {
        #pragma unroll
        for (int t = 0; t < 4; ++t) O[qs][t] = (v8f){};
        #pragma unroll
        for (int r = 0; r < 8; ++r) { mrow[qs][r] = -3.0e38f; lsum[qs][r] = 0.0f; }
    }

    const unsigned char* mrow_mask = mask + (size_t)b * MLA_L;
    const uint16_t* vbase = vt + (size_t)(b * MLA_H + h) * MLA_DH * MLA_L;

    for (int kbase = 0; kbase < MLA_L; kbase += 32) {
        // ---- K operand tiles (shared by both query sub-tiles)
        B32x8 kb0[2], kb1[2];
        {
            const uint16_t* kr0 = kb + ((size_t)(b * MLA_L + kbase + lm)      * MLA_D + h * MLA_DH);
            const uint16_t* kr1 = kb + ((size_t)(b * MLA_L + kbase + 16 + lm) * MLA_D + h * MLA_DH);
            #pragma unroll
            for (int t = 0; t < 2; ++t) {
                kb0[t].lo = *(const uint4*)(kr0 + t * 32 + half * 8);
                kb0[t].hi = *(const uint4*)(kr0 + t * 32 + 16 + half * 8);
                kb1[t].lo = *(const uint4*)(kr1 + t * 32 + half * 8);
                kb1[t].hi = *(const uint4*)(kr1 + t * 32 + 16 + half * 8);
            }
        }

        const float ok0 = mrow_mask[kbase + lm]      ? 0.0f : 1.0f;
        const float ok1 = mrow_mask[kbase + 16 + lm] ? 0.0f : 1.0f;

        #pragma unroll
        for (int qs = 0; qs < ATT_QS; ++qs) {
            // ---- S = Q @ K^T (two 16x16 tiles: keys kbase..+15, +16..+31)
            v8f S0 = {}, S1 = {};
            #pragma unroll
            for (int t = 0; t < 2; ++t) {
                S0 = wmma_bf16(qa[qs][t], kb0[t], S0);
                S1 = wmma_bf16(qa[qs][t], kb1[t], S1);
            }

            // ---- scale + key-padding mask (column = this lane's key)
            float s0r[8], s1r[8];
            #pragma unroll
            for (int r = 0; r < 8; ++r) {
                s0r[r] = (ok0 == 0.0f) ? S0[r] * sc : NEG;
                s1r[r] = (ok1 == 0.0f) ? S1[r] * sc : NEG;
            }

            // ---- online softmax (rows live on vgpr index within each half)
            uint16_t* pw = &Pt[wave][qs][0];
            #pragma unroll
            for (int r = 0; r < 8; ++r) {
                float mc = half16_max(fmaxf(s0r[r], s1r[r]));
                float mn = fmaxf(mrow[qs][r], mc);
                float rs = __expf(mrow[qs][r] - mn);
                mrow[qs][r] = mn;
                float p0 = __expf(s0r[r] - mn);
                float p1 = __expf(s1r[r] - mn);
                lsum[qs][r] = lsum[qs][r] * rs + half16_sum(p0 + p1);
                #pragma unroll
                for (int t = 0; t < 4; ++t) O[qs][t][r] *= rs;
                int row = r + 8 * half;                  // C->A layout bounce via LDS
                pw[row * 32 + lm]      = f2bf(p0);
                pw[row * 32 + 16 + lm] = f2bf(p1);
            }
        }

        asm volatile("s_wait_dscnt 0" ::: "memory");     // DS stores done before reload

        // ---- V operand tiles (shared by both query sub-tiles)
        B32x8 vb[4];
        #pragma unroll
        for (int t = 0; t < 4; ++t) {
            const uint16_t* vcol = vbase + (size_t)(t * 16 + lm) * MLA_L + kbase;
            vb[t].lo = *(const uint4*)(vcol + half * 8);
            vb[t].hi = *(const uint4*)(vcol + 16 + half * 8);
        }

        // ---- O += P @ V
        #pragma unroll
        for (int qs = 0; qs < ATT_QS; ++qs) {
            const uint16_t* pw = &Pt[wave][qs][0];
            B32x8 pa;
            pa.lo = *(const uint4*)(pw + lm * 32 + half * 8);
            pa.hi = *(const uint4*)(pw + lm * 32 + 16 + half * 8);
            #pragma unroll
            for (int t = 0; t < 4; ++t)
                O[qs][t] = wmma_bf16(pa, vb[t], O[qs][t]);
        }
    }

    // ---- epilogue: normalize and write bf16 [B*L, D]
    #pragma unroll
    for (int qs = 0; qs < ATT_QS; ++qs)
        #pragma unroll
        for (int t = 0; t < 4; ++t)
            #pragma unroll
            for (int r = 0; r < 8; ++r) {
                int m = qs * 16 + r + 8 * half;
                float o = O[qs][t][r] / lsum[qs][r];
                attn[(size_t)(b * MLA_L + q0 + m) * MLA_D + h * MLA_DH + t * 16 + lm] = f2bf(o);
            }
}

// ---------------------------------------------------------------------------
extern "C" void kernel_launch(void* const* d_in, const int* in_sizes, int n_in,
                              void* d_out, int out_size, void* d_ws, size_t ws_size,
                              hipStream_t stream) {
    const float*         x    = (const float*)d_in[0];
    const unsigned char* mask = (const unsigned char*)d_in[1];
    const float* Wdq  = (const float*)d_in[2];
    const float* bdq  = (const float*)d_in[3];
    const float* Wdkv = (const float*)d_in[4];
    const float* bdkv = (const float*)d_in[5];
    const float* Wuq  = (const float*)d_in[6];
    const float* buq  = (const float*)d_in[7];
    const float* Wuk  = (const float*)d_in[8];
    const float* buk  = (const float*)d_in[9];
    const float* Wuv  = (const float*)d_in[10];
    const float* buv  = (const float*)d_in[11];
    const float* Wo   = (const float*)d_in[12];
    const float* bo   = (const float*)d_in[13];
    float* out = (float*)d_out;

    const int BL  = MLA_B * MLA_L;          // 4096
    const int BLD = BL * MLA_D;             // 4,194,304

    // workspace allocator (bf16 buffers, 256B aligned)
    size_t off = 0;
    auto alloc = [&](size_t elems) -> uint16_t* {
        uint16_t* p = (uint16_t*)((char*)d_ws + off);
        off = (off + elems * 2 + 255) & ~(size_t)255;
        return p;
    };
    uint16_t* xbf    = alloc((size_t)BLD);
    uint16_t* wdq_t  = alloc((size_t)MLA_D * MLA_DC);
    uint16_t* wdkv_t = alloc((size_t)MLA_D * MLA_DC);
    uint16_t* wuq_t  = alloc((size_t)MLA_DC * MLA_D);
    uint16_t* wuk_t  = alloc((size_t)MLA_DC * MLA_D);
    uint16_t* wuv_t  = alloc((size_t)MLA_DC * MLA_D);
    uint16_t* wo_t   = alloc((size_t)MLA_D * MLA_D);
    uint16_t* dq_bf  = alloc((size_t)BL * MLA_DC);
    uint16_t* dkv_bf = alloc((size_t)BL * MLA_DC);
    uint16_t* q_bf   = alloc((size_t)BLD);
    uint16_t* k_bf   = alloc((size_t)BLD);
    uint16_t* v_bf   = alloc((size_t)BLD);
    uint16_t* vt_bf  = alloc((size_t)BLD);
    uint16_t* at_bf  = alloc((size_t)BLD);

    // 1) casts
    k_cast_bf16<<<(BLD + 255) / 256, 256, 0, stream>>>(x, xbf, BLD);
    k_cast_transpose_w<<<(MLA_D * MLA_DC + 255) / 256, 256, 0, stream>>>(Wdq,  wdq_t,  MLA_D,  MLA_DC);
    k_cast_transpose_w<<<(MLA_D * MLA_DC + 255) / 256, 256, 0, stream>>>(Wdkv, wdkv_t, MLA_D,  MLA_DC);
    k_cast_transpose_w<<<(MLA_DC * MLA_D + 255) / 256, 256, 0, stream>>>(Wuq,  wuq_t,  MLA_DC, MLA_D);
    k_cast_transpose_w<<<(MLA_DC * MLA_D + 255) / 256, 256, 0, stream>>>(Wuk,  wuk_t,  MLA_DC, MLA_D);
    k_cast_transpose_w<<<(MLA_DC * MLA_D + 255) / 256, 256, 0, stream>>>(Wuv,  wuv_t,  MLA_DC, MLA_D);
    k_cast_transpose_w<<<(MLA_D * MLA_D + 255) / 256, 256, 0, stream>>>(Wo,   wo_t,   MLA_D,  MLA_D);

    auto gemm = [&](const uint16_t* A, const uint16_t* Bt, const float* bias,
                    void* C, int M, int N, int K, int f32out) {
        int tiles = (M / (16 * GEMM_MT)) * (N / (16 * GEMM_NT));
        int blocks = (tiles + GEMM_WAVES - 1) / GEMM_WAVES;
        k_gemm_bf16<<<blocks, 32 * GEMM_WAVES, 0, stream>>>(A, Bt, bias, C, M, N, K, f32out);
    };

    // 2) latent down-projections
    gemm(xbf,    wdq_t,  bdq,  dq_bf,  BL, MLA_DC, MLA_D, 0);
    gemm(xbf,    wdkv_t, bdkv, dkv_bf, BL, MLA_DC, MLA_D, 0);
    // 3) up-projections
    gemm(dq_bf,  wuq_t,  buq,  q_bf,   BL, MLA_D, MLA_DC, 0);
    gemm(dkv_bf, wuk_t,  buk,  k_bf,   BL, MLA_D, MLA_DC, 0);
    gemm(dkv_bf, wuv_t,  buv,  v_bf,   BL, MLA_D, MLA_DC, 0);
    // 4) V transpose for attention B-operand
    k_transpose_v<<<(BLD + 255) / 256, 256, 0, stream>>>(v_bf, vt_bf);
    // 5) flash attention
    {
        int qwaves = MLA_B * MLA_H * (MLA_L / (16 * ATT_QS));  // 2048
        k_mla_attention<<<qwaves / ATT_WAVES, 32 * ATT_WAVES, 0, stream>>>(
            q_bf, k_bf, vt_bf, mask, at_bf);
    }
    // 6) output projection -> f32
    gemm(at_bf, wo_t, bo, out, BL, MLA_D, MLA_D, 1);

    (void)in_sizes; (void)n_in; (void)out_size; (void)ws_size;
}
